// DenseDotAttn_64020782514821
// MI455X (gfx1250) — compile-verified, matrix operand order checked by reading, not run on hardware
//
#include <hip/hip_runtime.h>

typedef __attribute__((ext_vector_type(4)))  float  v4f;
typedef __attribute__((ext_vector_type(8)))  float  v8f;
typedef __attribute__((ext_vector_type(8)))  __bf16 bf16x8;
typedef __attribute__((ext_vector_type(16))) __bf16 bf16x16;

#define WMMA_BF16(A, B, C) \
  __builtin_amdgcn_wmma_f32_16x16x32_bf16(false, (A), false, (B), (short)0, (C), false, false)

constexpr int Bn = 4;      // batch
constexpr int Mn = 2048;   // sequence
constexpr int Dm = 1024;   // d_model
constexpr int Hn = 16;     // heads
constexpr int Dh = 64;     // head dim

static __device__ inline bf16x16 cat8(bf16x8 a, bf16x8 b) {
  bf16x16 r;
#pragma unroll
  for (int i = 0; i < 8; ++i) { r[i] = a[i]; r[8 + i] = b[i]; }
  return r;
}

static __device__ inline v8f zero8() {
  v8f z;
#pragma unroll
  for (int i = 0; i < 8; ++i) z[i] = 0.0f;
  return z;
}

// ---------------------------------------------------------------------------
// Kernel 1: QKV projection.  Q,K stored bf16 [B,H,M,64]; V stored transposed
// bf16 [B,H,64,M] so the attention kernel's B-fragments are contiguous loads.
// W tiles staged in LDS *transposed* (n-major, padded rows) so each lane's
// B-fragment is two 16B ds_load_b128 instead of 16 scalar ds_load_u16.
// Block: 256 threads (8 waves), 128 rows of x per block, one head per block.y.
// ---------------------------------------------------------------------------
constexpr int WPAD = 40;   // 32 k-values + 8 pad: 80B rows (16B aligned, bank stride 20)

__global__ __launch_bounds__(256)
void qkv_proj_kernel(const float* __restrict__ x,
                     const float* __restrict__ wq,
                     const float* __restrict__ wk,
                     const float* __restrict__ wv,
                     unsigned short* __restrict__ Qo,
                     unsigned short* __restrict__ Ko,
                     unsigned short* __restrict__ Vto) {
  __shared__ __align__(16) __bf16 lwT[3][64][WPAD];  // [sel][n][k] transposed tiles
  __bf16* Q  = (__bf16*)Qo;
  __bf16* K  = (__bf16*)Ko;
  __bf16* Vt = (__bf16*)Vto;

  const int h    = blockIdx.y;
  const int tid  = threadIdx.x;
  const int wave = tid >> 5;
  const int lane = tid & 31;
  const int l15  = lane & 15;
  const int half = lane >> 4;

  const int rowBase = blockIdx.x * 128 + wave * 16;  // flattened b*M + m
  const int b  = rowBase >> 11;                      // / Mn
  const int m0 = rowBase & (Mn - 1);
  const size_t bh = (size_t)b * Hn + h;

  const float* w0 = wq + (size_t)h * Dm * Dh;
  const float* w1 = wk + (size_t)h * Dm * Dh;
  const float* w2 = wv + (size_t)h * Dm * Dh;

  v8f accQ[4], accK[4], accV[4];
#pragma unroll
  for (int t = 0; t < 4; ++t) { accQ[t] = zero8(); accK[t] = zero8(); accV[t] = zero8(); }

#pragma unroll 1
  for (int kk = 0; kk < Dm; kk += 32) {
    __syncthreads();
    // Stage 32x64 tiles of W_Q/W_K/W_V transposed into LDS (coalesced global
    // reads; scattered bf16 LDS stores are cheap and conflict-mitigated by pad)
#pragma unroll
    for (int it = 0; it < 8; ++it) {
      int i = tid + 256 * it;          // 0..2047
      int r = i >> 6, c = i & 63;      // r = k index, c = n index
      lwT[0][c][r] = (__bf16)w0[(size_t)(kk + r) * Dh + c];
      lwT[1][c][r] = (__bf16)w1[(size_t)(kk + r) * Dh + c];
      lwT[2][c][r] = (__bf16)w2[(size_t)(kk + r) * Dh + c];
    }
    __syncthreads();

    // A fragment from x: lane holds row (l15), k = half*8+j and 16+half*8+j
    const float* xr = x + (size_t)(rowBase + l15) * Dm + kk + half * 8;
    v4f x0 = *(const v4f*)(xr);
    v4f x1 = *(const v4f*)(xr + 4);
    v4f x2 = *(const v4f*)(xr + 16);
    v4f x3 = *(const v4f*)(xr + 20);
    bf16x16 aX;
#pragma unroll
    for (int j = 0; j < 4; ++j) {
      aX[j]      = (__bf16)x0[j];
      aX[4 + j]  = (__bf16)x1[j];
      aX[8 + j]  = (__bf16)x2[j];
      aX[12 + j] = (__bf16)x3[j];
    }

    auto mm = [&](v8f(&acc)[4], int sel) {
#pragma unroll
      for (int t = 0; t < 4; ++t) {
        int n = t * 16 + l15;
        const __bf16* wp = &lwT[sel][n][half * 8];   // 16B aligned
        bf16x16 bW = cat8(*(const bf16x8*)wp, *(const bf16x8*)(wp + 16));
        acc[t] = WMMA_BF16(aX, bW, acc[t]);
      }
    };
    mm(accQ, 0); mm(accK, 1); mm(accV, 2);
  }

  // Store: C layout is N=l15 (col), M = half*8 + v (row within 16-row tile)
#pragma unroll
  for (int t = 0; t < 4; ++t) {
    int col = t * 16 + l15;
#pragma unroll
    for (int v = 0; v < 8; ++v) {
      int m = m0 + half * 8 + v;
      Q[(bh * Mn + m) * Dh + col] = (__bf16)accQ[t][v];
      K[(bh * Mn + m) * Dh + col] = (__bf16)accK[t][v];
    }
    bf16x8 vv;
#pragma unroll
    for (int v = 0; v < 8; ++v) vv[v] = (__bf16)accV[t][v];
    *(bf16x8*)(Vt + (bh * Dh + col) * (size_t)Mn + m0 + half * 8) = vv;  // contiguous 16B
  }
}

// ---------------------------------------------------------------------------
// Kernel 2: flash attention.  One (b,h) per block.y, 128 query rows per block
// (16 per wave).  32-key chunks: 4 WMMAs for S=Q*K^T, online softmax with
// wave32 shfl reductions, P through per-wave LDS for layout swap, 4 WMMAs for
// O += P*V.  Output fp32 directly into [B,M,H*64].
// ---------------------------------------------------------------------------
__global__ __launch_bounds__(256)
void attn_kernel(const unsigned short* __restrict__ Qi,
                 const unsigned short* __restrict__ Ki,
                 const unsigned short* __restrict__ Vti,
                 float* __restrict__ out) {
  __shared__ __align__(16) __bf16 pl[8][16][32];   // per-wave P tile
  const __bf16* Q  = (const __bf16*)Qi;
  const __bf16* K  = (const __bf16*)Ki;
  const __bf16* Vt = (const __bf16*)Vti;

  const int tid  = threadIdx.x;
  const int wave = tid >> 5;
  const int lane = tid & 31;
  const int l15  = lane & 15;
  const int half = lane >> 4;

  const int bh = blockIdx.y;
  const int b  = bh >> 4;
  const int h  = bh & 15;
  const int q0 = blockIdx.x * 128 + wave * 16;

  const size_t qkBase = (size_t)bh * Mn * Dh;
  const size_t vtBase = (size_t)bh * Dh * Mn;

  // Q A-fragments, loaded once (d 0..31, d 32..63)
  bf16x16 aQ[2];
  {
    const __bf16* qr = Q + qkBase + (size_t)(q0 + l15) * Dh + half * 8;
#pragma unroll
    for (int f = 0; f < 2; ++f)
      aQ[f] = cat8(*(const bf16x8*)(qr + f * 32), *(const bf16x8*)(qr + f * 32 + 16));
  }

  v8f accO[4];
#pragma unroll
  for (int t = 0; t < 4; ++t) accO[t] = zero8();
  float mrow[8], lrow[8];
#pragma unroll
  for (int v = 0; v < 8; ++v) { mrow[v] = -3.0e38f; lrow[v] = 0.0f; }

  const float scale = 0.125f;  // 1/sqrt(64)

#pragma unroll 1
  for (int c = 0; c < Mn; c += 32) {
    // S = Q * K^T for keys [c, c+32): two 16-key subtiles, K-dim 64 = 2 WMMAs
    v8f s[2];
#pragma unroll
    for (int sub = 0; sub < 2; ++sub) {
      const __bf16* kr = K + qkBase + (size_t)(c + sub * 16 + l15) * Dh + half * 8;
      bf16x16 bK0 = cat8(*(const bf16x8*)(kr),      *(const bf16x8*)(kr + 16));
      bf16x16 bK1 = cat8(*(const bf16x8*)(kr + 32), *(const bf16x8*)(kr + 48));
      v8f z = zero8();
      z = WMMA_BF16(aQ[0], bK0, z);
      z = WMMA_BF16(aQ[1], bK1, z);
      s[sub] = z;
    }

    // Prefetch next chunk of K / V^T while we do softmax math
    if (c + 32 < Mn) {
      __builtin_prefetch(K + qkBase + (size_t)(c + 32 + l15) * Dh, 0, 1);
      __builtin_prefetch(Vt + vtBase + (size_t)(l15 * 4) * Mn + c + 32, 0, 1);
    }

    // Online softmax: per-row stats via 16-lane xor-shuffle reductions
    float al[8];
#pragma unroll
    for (int v = 0; v < 8; ++v) {
      float s0 = s[0][v] * scale;
      float s1 = s[1][v] * scale;
      float cm = fmaxf(s0, s1);
#pragma unroll
      for (int mk = 1; mk < 16; mk <<= 1) cm = fmaxf(cm, __shfl_xor(cm, mk, 32));
      float nmax  = fmaxf(mrow[v], cm);
      float alpha = __expf(mrow[v] - nmax);
      float p0 = __expf(s0 - nmax);
      float p1 = __expf(s1 - nmax);
      float ps = p0 + p1;
#pragma unroll
      for (int mk = 1; mk < 16; mk <<= 1) ps += __shfl_xor(ps, mk, 32);
      lrow[v] = lrow[v] * alpha + ps;
      mrow[v] = nmax;
      al[v]   = alpha;
      int r = half * 8 + v;                 // C-layout row -> LDS row
      pl[wave][r][l15]      = (__bf16)p0;
      pl[wave][r][16 + l15] = (__bf16)p1;
    }
#pragma unroll
    for (int t = 0; t < 4; ++t)
#pragma unroll
      for (int v = 0; v < 8; ++v) accO[t][v] *= al[v];

    __syncthreads();
    // Re-read P in A-fragment layout: row = l15, k = half*8+j (+16)
    const __bf16* pr = &pl[wave][l15][half * 8];
    bf16x16 aP = cat8(*(const bf16x8*)pr, *(const bf16x8*)(pr + 16));
    __syncthreads();

    // O += P * V  (V^T layout makes B-fragment loads contiguous)
#pragma unroll
    for (int t = 0; t < 4; ++t) {
      const __bf16* vr = Vt + vtBase + (size_t)(t * 16 + l15) * Mn + c + half * 8;
      bf16x16 bV = cat8(*(const bf16x8*)vr, *(const bf16x8*)(vr + 16));
      accO[t] = WMMA_BF16(aP, bV, accO[t]);
    }
  }

  // Epilogue: normalize and write out[b][m][h*64 + dv] (fp32)
  float* ob = out + (size_t)b * Mn * Dm + (size_t)h * Dh;
#pragma unroll
  for (int v = 0; v < 8; ++v) {
    float inv = 1.0f / lrow[v];
    int m = q0 + half * 8 + v;
#pragma unroll
    for (int t = 0; t < 4; ++t)
      ob[(size_t)m * Dm + t * 16 + l15] = accO[t][v] * inv;
  }
}

// ---------------------------------------------------------------------------
extern "C" void kernel_launch(void* const* d_in, const int* in_sizes, int n_in,
                              void* d_out, int out_size, void* d_ws, size_t ws_size,
                              hipStream_t stream) {
  (void)in_sizes; (void)n_in; (void)out_size; (void)ws_size;
  const float* x  = (const float*)d_in[0];
  const float* wq = (const float*)d_in[1];
  const float* wk = (const float*)d_in[2];
  const float* wv = (const float*)d_in[3];
  float* out = (float*)d_out;

  const size_t SZ = (size_t)Bn * Hn * Mn * Dh;      // 8.39M elems (bf16) each
  unsigned short* Qb  = (unsigned short*)d_ws;
  unsigned short* Kb  = Qb + SZ;
  unsigned short* Vtb = Kb + SZ;

  dim3 gProj((Bn * Mn) / 128, Hn);                  // 64 x 16 blocks
  qkv_proj_kernel<<<gProj, 256, 0, stream>>>(x, wq, wk, wv, Qb, Kb, Vtb);

  dim3 gAttn(Mn / 128, Bn * Hn);                    // 16 x 64 blocks
  attn_kernel<<<gAttn, 256, 0, stream>>>(Qb, Kb, Vtb, out);
}
